// Q_AttnFFN_72859825209931
// MI455X (gfx1250) — matrix operand order repaired and checked by reading, not compiled, not run on hardware
//
#include <hip/hip_runtime.h>

typedef __attribute__((ext_vector_type(16))) _Float16 v16h;
typedef __attribute__((ext_vector_type(8)))  _Float16 v8h;
typedef __attribute__((ext_vector_type(8)))  float    v8f;

#define BATCH  512
#define CIN    384
#define NSP    49
#define NP     64
#define NHEADS 8
#define KD     32
#define DHEAD  128
#define NHKD   256
#define DHTOT  1024
#define HIDD   1536

enum {
  S_Q = 0, S_K, S_V, S_VL, S_PRETH1, S_ATTN, S_POST, S_AVRAW, S_AVRELU,
  S_PROJ, S_X1, S_H1, S_MID, S_F2, S_OUT
};

// ---------------- small device helpers ----------------

__device__ __forceinline__ float sc8(const float* SC, int i) {
  return fmaxf(SC[i] * (1.0f / 127.0f), 1e-8f);
}
__device__ __forceinline__ float fq8(float v, float s) {
  return fminf(fmaxf(rintf(v / s), -127.0f), 127.0f) * s;   // round-half-even == jnp.round
}
__device__ __forceinline__ float fq8i(float v, float s, float inv) {
  return fminf(fmaxf(rintf(v * inv), -127.0f), 127.0f) * s;
}
__device__ __forceinline__ void atomic_absmax(float* SC, int idx, float v) {
  #pragma unroll
  for (int o = 16; o > 0; o >>= 1) v = fmaxf(v, __shfl_xor(v, o, 32));
  if ((threadIdx.x & 31) == 0)
    atomicMax(reinterpret_cast<unsigned int*>(&SC[idx]), __float_as_uint(v));
}

// ---------------- WMMA fragment helpers (wave32, 16x16x32 f16) ----------------
// All operands live "K-contiguous": A row m / B col n each hold K halves contiguously.

__device__ __forceinline__ v16h concat8(v8h a, v8h b) {
  return __builtin_shufflevector(a, b, 0,1,2,3,4,5,6,7,8,9,10,11,12,13,14,15);
}

// A 16(M)x32(K): lane l<16 -> M=l, elems = K 0..15 ; lanes 16..31 -> K 16..31
__device__ __forceinline__ v16h frag_A_raw(const _Float16* __restrict__ base, int stride, int k0) {
  const int lane = threadIdx.x & 31;
  const int m = lane & 15, hi = lane >> 4;
  const _Float16* p = base + (size_t)m * stride + k0 + hi * 16;
  return concat8(*(const v8h*)p, *(const v8h*)(p + 8));
}
__device__ __forceinline__ v16h frag_A_q(const _Float16* __restrict__ base, int stride, int k0,
                                         float s, float inv, int kvalid) {
  const int lane = threadIdx.x & 31;
  const int m = lane & 15, hi = lane >> 4;
  const _Float16* p = base + (size_t)m * stride + k0 + hi * 16;
  v8h a = *(const v8h*)p;
  v8h b = *(const v8h*)(p + 8);
  v16h r;
  #pragma unroll
  for (int e = 0; e < 8; ++e) {
    int ka = k0 + hi * 16 + e;
    r[e]     = (_Float16)((ka      < kvalid) ? fq8i((float)a[e], s, inv) : 0.0f);
    r[8 + e] = (_Float16)((ka + 8  < kvalid) ? fq8i((float)b[e], s, inv) : 0.0f);
  }
  return r;
}
// B 32(K)x16(N): lane l<16 -> N=l, elems = K 0..15 ; lanes 16..31 -> K 16..31
__device__ __forceinline__ v16h frag_B_q(const _Float16* __restrict__ base, int stride, int k0,
                                         float s, float inv) {
  const int lane = threadIdx.x & 31;
  const int n = lane & 15, koff = (lane >> 4) * 16;
  const _Float16* p = base + (size_t)n * stride + k0 + koff;
  v8h a = *(const v8h*)p;
  v8h b = *(const v8h*)(p + 8);
  v16h r;
  #pragma unroll
  for (int e = 0; e < 8; ++e) {
    r[e]     = (_Float16)fq8i((float)a[e], s, inv);
    r[8 + e] = (_Float16)fq8i((float)b[e], s, inv);
  }
  return r;
}

// ---------------- setup kernels ----------------

__global__ void k_init(float* SC) {
  if (threadIdx.x < 64) SC[threadIdx.x] = 0.0f;
}

__global__ __launch_bounds__(256)
void k_quant_w_f16(const float* __restrict__ w, _Float16* __restrict__ dst, int K) {
  __shared__ float red[256];
  const int o = blockIdx.x;
  float m = 0.0f;
  for (int i = threadIdx.x; i < K; i += 256) m = fmaxf(m, fabsf(w[(size_t)o * K + i]));
  red[threadIdx.x] = m; __syncthreads();
  for (int s = 128; s > 0; s >>= 1) {
    if (threadIdx.x < s) red[threadIdx.x] = fmaxf(red[threadIdx.x], red[threadIdx.x + s]);
    __syncthreads();
  }
  const float sc = fmaxf(red[0] * (1.0f / 127.0f), 1e-8f);
  for (int i = threadIdx.x; i < K; i += 256) {
    float v = w[(size_t)o * K + i];
    v = fminf(fmaxf(rintf(v / sc), -127.0f), 127.0f) * sc;
    dst[(size_t)o * K + i] = (_Float16)v;
  }
}

__global__ __launch_bounds__(256)
void k_quant_w_f32(const float* __restrict__ w, float* __restrict__ dst, int K) {
  __shared__ float red[256];
  const int o = blockIdx.x;
  float m = 0.0f;
  for (int i = threadIdx.x; i < K; i += 256) m = fmaxf(m, fabsf(w[(size_t)o * K + i]));
  red[threadIdx.x] = m; __syncthreads();
  for (int s = 128; s > 0; s >>= 1) {
    if (threadIdx.x < s) red[threadIdx.x] = fmaxf(red[threadIdx.x], red[threadIdx.x + s]);
    __syncthreads();
  }
  const float sc = fmaxf(red[0] * (1.0f / 127.0f), 1e-8f);
  for (int i = threadIdx.x; i < K; i += 256) {
    float v = w[(size_t)o * K + i];
    dst[(size_t)o * K + i] = fminf(fmaxf(rintf(v / sc), -127.0f), 127.0f) * sc;
  }
}

__global__ void k_concat_bias(const float* bq, const float* bk, const float* bv, float* dst) {
  int i = blockIdx.x * 256 + threadIdx.x;
  if (i < 256)       dst[i] = bq[i];
  else if (i < 512)  dst[i] = bk[i - 256];
  else if (i < 1536) dst[i] = bv[i - 512];
}

// x (b,c,49) f32 -> XH (b,64,384) f16 n-major, zero pad rows
__global__ __launch_bounds__(256)
void k_pack_x(const float* __restrict__ x, _Float16* __restrict__ XH) {
  int idx = blockIdx.x * 256 + threadIdx.x;          // == (b*64+n)*384+c
  int c = idx % CIN; int n = (idx / CIN) % NP; int b = idx / (CIN * NP);
  float v = (n < NSP) ? x[((size_t)b * CIN + c) * NSP + n] : 0.0f;
  XH[idx] = (_Float16)v;
}

// ---------------- generic per-sample GEMM: out[b,n,o] = sum_k W[o,k] X[b,n,k] ----------------
// 128 thr = 4 waves; wave w owns output channels [o0+16w, o0+16w+16); A direct from global,
// B (activation rows) staged once per k-step in LDS, shared by the 4 waves.
// Non-quantizing inputs use GLOBAL_LOAD_ASYNC_TO_LDS_B128 (ASYNCcnt) for the staging copy.
__global__ __launch_bounds__(128)
void k_gemm(const _Float16* __restrict__ W, const _Float16* __restrict__ Xact,
            const float* __restrict__ bias, _Float16* __restrict__ Yn,
            _Float16* __restrict__ Yv, float* __restrict__ SC,
            int O, int K, int ostride, int x_q_idx, int absmax_mode, int do_relu)
{
  __shared__ _Float16 Bs[64 * 40];
  const int b = blockIdx.y, o0 = blockIdx.x * 64;
  const int tid = threadIdx.x, wave = tid >> 5, lane = tid & 31;
  const _Float16* Xb = Xact + (size_t)b * NP * K;
  const bool qin = (x_q_idx >= 0);
  const float s_in = qin ? sc8(SC, x_q_idx) : 1.0f;
  const float inv_in = 1.0f / s_in;
  const _Float16* Wrow = W + (size_t)(o0 + wave * 16) * K;

  v8f acc[4];
  #pragma unroll
  for (int i = 0; i < 4; ++i) acc[i] = 0;

  for (int k0 = 0; k0 < K; k0 += 32) {
    if (qin) {                                    // fake-quant between global and LDS
      #pragma unroll
      for (int t = 0; t < 2; ++t) {
        int j = tid + t * 128;                    // 0..255 chunks of 8 halves
        int n = j >> 2, c8 = (j & 3) * 8;
        v8h v = *(const v8h*)(Xb + (size_t)n * K + k0 + c8);
        #pragma unroll
        for (int e = 0; e < 8; ++e) v[e] = (_Float16)fq8i((float)v[e], s_in, inv_in);
        *(v8h*)(&Bs[n * 40 + c8]) = v;
      }
    } else {                                      // raw copy: async DMA into LDS
      #pragma unroll
      for (int t = 0; t < 2; ++t) {
        int j = tid + t * 128;
        int n = j >> 2, c8 = (j & 3) * 8;
        unsigned lds_off = (unsigned)(size_t)(&Bs[n * 40 + c8]);
        const _Float16* g = Xb + (size_t)n * K + k0 + c8;
        asm volatile("global_load_async_to_lds_b128 %0, %1, off"
                     :: "v"(lds_off), "v"(g) : "memory");
      }
      asm volatile("s_wait_asynccnt 0x0" ::: "memory");
    }
    if (k0 + 32 < K) __builtin_prefetch(Wrow + (size_t)(lane & 15) * K + k0 + 32, 0, 1);
    __syncthreads();
    v16h a = frag_A_raw(Wrow, K, k0);
    #pragma unroll
    for (int nt = 0; nt < 4; ++nt) {
      const _Float16* p = &Bs[(nt * 16 + (lane & 15)) * 40 + (lane >> 4) * 16];
      v16h bf = concat8(*(const v8h*)p, *(const v8h*)(p + 8));
      acc[nt] = __builtin_amdgcn_wmma_f32_16x16x32_f16(false, a, false, bf,
                                                       (short)0, acc[nt], false, false);
    }
    __syncthreads();
  }

  const int col = lane & 15, rbase = (lane >> 4) * 8;
  const int mbase = wave * 16 + rbase;
  float vmax = 0.0f;
  #pragma unroll
  for (int nt = 0; nt < 4; ++nt) {
    int n = nt * 16 + col;
    v8h pack;
    #pragma unroll
    for (int r = 0; r < 8; ++r) {
      float v = acc[nt][r] + bias[o0 + mbase + r];
      if (do_relu) v = fmaxf(v, 0.0f);
      if (n < NSP) vmax = fmaxf(vmax, fabsf(v));
      pack[r] = (_Float16)v;
    }
    if (Yv != nullptr && o0 >= 512) {             // v-part: channel-major (b,1024,64)
      int ch = o0 - 512 + mbase;
      #pragma unroll
      for (int r = 0; r < 8; ++r)
        Yv[((size_t)b * DHTOT + ch + r) * NP + n] = pack[r];
    } else {                                      // n-major (b,64,ostride), 16B store
      *(v8h*)(Yn + ((size_t)b * NP + n) * ostride + o0 + mbase) = pack;
    }
  }
  if (absmax_mode >= 0) atomic_absmax(SC, absmax_mode, vmax);
  else if (absmax_mode == -2) {
    int idx = (o0 < 256) ? S_Q : (o0 < 512 ? S_K : S_V);
    atomic_absmax(SC, idx, vmax);
  }
}

// ---------------- attn = q k^T * scale + bias  (per (b,h); q,k n-major -> LDS-free) ----------------
__global__ __launch_bounds__(128)
void k_qk(const _Float16* __restrict__ QKN, const float* __restrict__ ab,
          const int* __restrict__ bidx, _Float16* __restrict__ ATTN,
          float* __restrict__ SC)
{
  const int bh = blockIdx.x, b = bh >> 3, h = bh & 7;
  const int tid = threadIdx.x, wave = tid >> 5, lane = tid & 31;
  const float sQ = sc8(SC, S_Q), sK = sc8(SC, S_K);
  const float iQ = 1.0f / sQ, iK = 1.0f / sK;
  const _Float16* Qbase = QKN + ((size_t)b * NP + wave * 16) * 512 + h * KD;   // rows n
  const _Float16* Kbase = QKN + ((size_t)b * NP) * 512 + NHKD + h * KD;        // rows m

  v8f acc[4];
  #pragma unroll
  for (int i = 0; i < 4; ++i) acc[i] = 0;
  v16h a = frag_A_q(Qbase, 512, 0, sQ, iQ, KD);
  #pragma unroll
  for (int nt = 0; nt < 4; ++nt) {
    v16h bf = frag_B_q(Kbase + (size_t)(nt * 16) * 512, 512, 0, sK, iK);
    acc[nt] = __builtin_amdgcn_wmma_f32_16x16x32_f16(false, a, false, bf,
                                                     (short)0, acc[nt], false, false);
  }
  const int col = lane & 15, rbase = (lane >> 4) * 8;
  float vmax = 0.0f;
  _Float16* Ao = ATTN + (size_t)bh * NP * NP;
  #pragma unroll
  for (int nt = 0; nt < 4; ++nt) {
    int m = nt * 16 + col;
    #pragma unroll
    for (int r = 0; r < 8; ++r) {
      int n = wave * 16 + rbase + r;
      float v = acc[nt][r] * 0.17677669529663687f;   // 32^-0.5
      if (n < NSP && m < NSP) {
        v += ab[h * NSP + bidx[n * NSP + m]];
        vmax = fmaxf(vmax, fabsf(v));
      }
      Ao[(size_t)n * NP + m] = (_Float16)v;
    }
  }
  atomic_absmax(SC, S_PRETH1, vmax);
}

// ---------------- talking heads: 8x8 mix over head axis ----------------
__global__ __launch_bounds__(256)
void k_headmix(const _Float16* __restrict__ IN, _Float16* __restrict__ OUT,
               const float* __restrict__ W8, const float* __restrict__ B8,
               float* __restrict__ SC, int s_in, int s_out)
{
  const int total = BATCH * NSP * NSP;
  int idx = blockIdx.x * 256 + threadIdx.x;
  float amax = 0.0f;
  if (idx < total) {
    int m = idx % NSP; int t = idx / NSP; int n = t % NSP; int b = t / NSP;
    float s = (s_in >= 0) ? sc8(SC, s_in) : 0.0f;
    float in[8];
    #pragma unroll
    for (int i = 0; i < 8; ++i) {
      float v = (float)IN[(((size_t)b * 8 + i) * NP + n) * NP + m];
      if (s_in >= 0) v = fq8(v, s);
      in[i] = v;
    }
    #pragma unroll
    for (int o = 0; o < 8; ++o) {
      float acc = B8[o];
      #pragma unroll
      for (int i = 0; i < 8; ++i) acc += W8[o * 8 + i] * in[i];
      amax = fmaxf(amax, fabsf(acc));
      OUT[(((size_t)b * 8 + o) * NP + n) * NP + m] = (_Float16)acc;
    }
  }
  atomic_absmax(SC, s_out, amax);
}

// ---------------- I-BERT integer softmax: one wave per (b,h,n) row of 49 ----------------
__device__ __forceinline__ float ibert_e2(float xi, float mx, float sf) {
  xi -= mx;
  float x0 = floorf(-0.6931f / sf);
  xi = fmaxf(xi, 30.0f * x0);
  float q = floorf(xi / x0);
  float r = xi - x0 * q;
  float bi = floorf(2.70732379f / sf);
  float ci = floorf(2.79211472f / (sf * sf));
  float z = (r + bi) * r + ci;
  float ei = fmaxf(floorf(z * exp2f(30.0f - q)), 0.0f);
  float fac = 0.35815147f * sf * sf * (1.0f / 1073741824.0f);
  float ev = ei * fac;
  float emax = ci * 1073741824.0f * fac;   // analytic tensor max (r=0,q=0 in every row)
  float s16 = fmaxf(emax * (1.0f / 32767.0f), 1e-8f);
  return fminf(fmaxf(rintf(ev / s16), -32767.0f), 32767.0f);
}

__global__ __launch_bounds__(256)
void k_softmax(const _Float16* __restrict__ IN, _Float16* __restrict__ OUT,
               const float* __restrict__ SC, int sidx)
{
  const int warp = threadIdx.x >> 5, lane = threadIdx.x & 31;
  const int row = blockIdx.x * 8 + warp;
  const int bh = row / NSP, n = row % NSP;
  const size_t base = ((size_t)bh * NP + n) * NP;
  const float sf = fmaxf(SC[sidx] * (1.0f / 127.0f), 1e-8f);
  const int m1 = lane, m2 = lane + 32;
  const bool v2 = (m2 < NSP);
  float x1 = fminf(fmaxf(rintf((float)IN[base + m1] / sf), -127.0f), 127.0f);
  float x2 = v2 ? fminf(fmaxf(rintf((float)IN[base + m2] / sf), -127.0f), 127.0f) : -3.0e38f;
  float mx = fmaxf(x1, x2);
  #pragma unroll
  for (int o = 16; o > 0; o >>= 1) mx = fmaxf(mx, __shfl_xor(mx, o, 32));
  float e1 = ibert_e2(x1, mx, sf);
  float e2 = v2 ? ibert_e2(x2, mx, sf) : 0.0f;
  float sum = e1 + e2;
  #pragma unroll
  for (int o = 16; o > 0; o >>= 1) sum += __shfl_xor(sum, o, 32);
  float factor = floorf(4294967296.0f / sum);
  float o1 = floorf(e1 * factor * (1.0f / 65536.0f)) * (1.0f / 65536.0f);
  float o2 = floorf(e2 * factor * (1.0f / 65536.0f)) * (1.0f / 65536.0f);
  OUT[base + m1] = (_Float16)o1;
  if (v2) OUT[base + m2] = (_Float16)o2;
}

// ---------------- out[b,n,d] = sum_m attn[n,m] v[m,d] + vl, via D[d,n] = v^T x attn^T ----------------
// v stored channel-major (b,1024,64): rows d are m-contiguous => direct A fragments.
// attn (b,h,n,m): rows n are m-contiguous => direct B fragments. LDS-free.
__global__ __launch_bounds__(128)
void k_av(const _Float16* __restrict__ ATTN2, const _Float16* __restrict__ VBUF,
          const _Float16* __restrict__ VL, _Float16* __restrict__ AV,
          float* __restrict__ SC)
{
  const int bh = blockIdx.x, b = bh >> 3, h = bh & 7;
  const int yhalf = blockIdx.y;                       // d block of 64 within the head
  const int tid = threadIdx.x, wave = tid >> 5, lane = tid & 31;
  const float sP = sc8(SC, S_POST), sV = sc8(SC, S_V), sVL = sc8(SC, S_VL);
  const float iP = 1.0f / sP, iV = 1.0f / sV, iVL = 1.0f / sVL;
  const int d0 = yhalf * 64 + wave * 16;
  const _Float16* Vbase = VBUF + ((size_t)b * DHTOT + h * DHEAD + d0) * NP;   // rows d
  const _Float16* Abase = ATTN2 + (size_t)bh * NP * NP;                        // rows n

  v8f acc[4];
  #pragma unroll
  for (int i = 0; i < 4; ++i) acc[i] = 0;
  #pragma unroll
  for (int k0 = 0; k0 < 64; k0 += 32) {
    v16h a = frag_A_q(Vbase, NP, k0, sV, iV, NSP);    // zero K=m >= 49
    #pragma unroll
    for (int nt = 0; nt < 4; ++nt) {
      v16h bf = frag_B_q(Abase + (size_t)(nt * 16) * NP, NP, k0, sP, iP);
      acc[nt] = __builtin_amdgcn_wmma_f32_16x16x32_f16(false, a, false, bf,
                                                       (short)0, acc[nt], false, false);
    }
  }
  const int col = lane & 15, rbase = (lane >> 4) * 8;
  const int ch0 = h * DHEAD + d0 + rbase;             // 8 contiguous output channels
  float vmax = 0.0f;
  #pragma unroll
  for (int nt = 0; nt < 4; ++nt) {
    int n = nt * 16 + col;
    v8h pack;
    if (n < NSP) {
      v8h vlv = *(const v8h*)(VL + ((size_t)b * NP + n) * DHTOT + ch0);
      #pragma unroll
      for (int r = 0; r < 8; ++r) {
        float v = acc[nt][r] + fq8i((float)vlv[r], sVL, iVL);
        vmax = fmaxf(vmax, fabsf(v));
        pack[r] = (_Float16)v;
      }
    } else {
      #pragma unroll
      for (int r = 0; r < 8; ++r) pack[r] = (_Float16)acc[nt][r];
    }
    *(v8h*)(AV + ((size_t)b * NP + n) * DHTOT + ch0) = pack;
  }
  atomic_absmax(SC, S_AVRAW, vmax);
}

// ---------------- depthwise 3x3 (pad=1) over 7x7 ----------------
// vl branch: input VBUF channel-major, output VL n-major
__global__ __launch_bounds__(256)
void k_dw_vl(const _Float16* __restrict__ VBUF, const float* __restrict__ W9,
             const float* __restrict__ bias, _Float16* __restrict__ VL,
             float* __restrict__ SC)
{
  const int total = BATCH * DHTOT * NSP;
  int idx = blockIdx.x * 256 + threadIdx.x;
  float amax = 0.0f;
  if (idx < total) {
    int n = idx % NSP; int ch = (idx / NSP) % DHTOT; int b = idx / (NSP * DHTOT);
    int y = n / 7, xx = n % 7;
    const float s = sc8(SC, S_V);
    const _Float16* src = VBUF + ((size_t)b * DHTOT + ch) * NP;
    float acc = bias[ch];
    #pragma unroll
    for (int ky = 0; ky < 3; ++ky) {
      int yy = y + ky - 1;
      if (yy < 0 || yy > 6) continue;
      #pragma unroll
      for (int kx = 0; kx < 3; ++kx) {
        int xc = xx + kx - 1;
        if (xc < 0 || xc > 6) continue;
        acc += W9[ch * 9 + ky * 3 + kx] * fq8((float)src[yy * 7 + xc], s);
      }
    }
    amax = fabsf(acc);
    VL[((size_t)b * NP + n) * DHTOT + ch] = (_Float16)acc;
  }
  atomic_absmax(SC, S_VL, amax);
}

// mid branch: input H1 n-major, output MID n-major, channel-fastest (coalesced)
__global__ __launch_bounds__(256)
void k_dw_mid(const _Float16* __restrict__ H1, const float* __restrict__ W9,
              const float* __restrict__ bias, _Float16* __restrict__ MID,
              float* __restrict__ SC)
{
  const int total = BATCH * HIDD * NSP;
  int idx = blockIdx.x * 256 + threadIdx.x;
  float amax = 0.0f;
  if (idx < total) {
    int ch = idx % HIDD; int n = (idx / HIDD) % NSP; int b = idx / (HIDD * NSP);
    int y = n / 7, xx = n % 7;
    const float s = sc8(SC, S_H1);
    const _Float16* src = H1 + (size_t)b * NP * HIDD + ch;
    float acc = bias[ch];
    #pragma unroll
    for (int ky = 0; ky < 3; ++ky) {
      int yy = y + ky - 1;
      if (yy < 0 || yy > 6) continue;
      #pragma unroll
      for (int kx = 0; kx < 3; ++kx) {
        int xc = xx + kx - 1;
        if (xc < 0 || xc > 6) continue;
        acc += W9[ch * 9 + ky * 3 + kx] * fq8((float)src[(size_t)(yy * 7 + xc) * HIDD], s);
      }
    }
    acc = fmaxf(acc, 0.0f);
    amax = acc;
    MID[((size_t)b * NP + n) * HIDD + ch] = (_Float16)acc;
  }
  atomic_absmax(SC, S_MID, amax);
}

// ---------------- elementwise glue ----------------

__global__ __launch_bounds__(256)
void k_relu_q(_Float16* __restrict__ AV, float* __restrict__ SC) {
  const int total = BATCH * DHTOT * NSP;
  int idx = blockIdx.x * 256 + threadIdx.x;
  float amax = 0.0f;
  if (idx < total) {
    int ch = idx % DHTOT; int n = (idx / DHTOT) % NSP; int b = idx / (DHTOT * NSP);
    size_t a = ((size_t)b * NP + n) * DHTOT + ch;
    float v = fq8((float)AV[a], sc8(SC, S_AVRAW));  // quant_av_res
    v = fmaxf(v, 0.0f);                              // proj_act
    amax = v;
    AV[a] = (_Float16)v;
  }
  atomic_absmax(SC, S_AVRELU, amax);
}

__global__ __launch_bounds__(256)
void k_resid_a(const float* __restrict__ x, const _Float16* __restrict__ PROJ,
               const float* __restrict__ ls1, _Float16* __restrict__ T1,
               float* __restrict__ SC) {
  const int total = BATCH * NSP * CIN;
  int idx = blockIdx.x * 256 + threadIdx.x;
  float amax = 0.0f;
  if (idx < total) {
    int c = idx % CIN; int n = (idx / CIN) % NSP; int b = idx / (CIN * NSP);
    float pj = fq8((float)PROJ[((size_t)b * NP + n) * CIN + c], sc8(SC, S_PROJ));
    float t = x[((size_t)b * CIN + c) * NSP + n] + ls1[c] * pj;
    amax = fabsf(t);
    T1[idx] = (_Float16)t;                           // (b,49,384)
  }
  atomic_absmax(SC, S_X1, amax);
}

__global__ __launch_bounds__(256)
void k_resid_b(const _Float16* __restrict__ T1, _Float16* __restrict__ X1,
               const float* __restrict__ SC) {
  int idx = blockIdx.x * 256 + threadIdx.x;          // over (b,64,384)
  int c = idx % CIN; int n = (idx / CIN) % NP; int b = idx / (CIN * NP);
  float v = 0.0f;
  if (n < NSP) v = fq8((float)T1[((size_t)b * NSP + n) * CIN + c], sc8(SC, S_X1));
  X1[idx] = (_Float16)v;
}

__global__ __launch_bounds__(256)
void k_final_a(const _Float16* __restrict__ X1, const _Float16* __restrict__ F2,
               const float* __restrict__ ls2, float* __restrict__ SC) {
  const int total = BATCH * NSP * CIN;
  int idx = blockIdx.x * 256 + threadIdx.x;
  float amax = 0.0f;
  if (idx < total) {
    int c = idx % CIN; int n = (idx / CIN) % NSP; int b = idx / (CIN * NSP);
    size_t ap = ((size_t)b * NP + n) * CIN + c;
    float t = (float)X1[ap] + ls2[c] * fq8((float)F2[ap], sc8(SC, S_F2));
    amax = fabsf(t);
  }
  atomic_absmax(SC, S_OUT, amax);
}

__global__ __launch_bounds__(256)
void k_final_b(const _Float16* __restrict__ X1, const _Float16* __restrict__ F2,
               const float* __restrict__ ls2, const float* __restrict__ SC,
               float* __restrict__ out) {
  const int total = BATCH * NSP * CIN;
  int idx = blockIdx.x * 256 + threadIdx.x;
  if (idx < total) {
    int c = idx % CIN; int n = (idx / CIN) % NSP; int b = idx / (CIN * NSP);
    size_t ap = ((size_t)b * NP + n) * CIN + c;
    float t = (float)X1[ap] + ls2[c] * fq8((float)F2[ap], sc8(SC, S_F2));
    out[((size_t)b * CIN + c) * NSP + n] = fq8(t, sc8(SC, S_OUT));
  }
  if (idx == 0) out[total] = fmaxf(SC[S_OUT] * (1.0f / 127.0f), 1e-8f);  // out_sf
}

// ---------------- host launcher ----------------

extern "C" void kernel_launch(void* const* d_in, const int* in_sizes, int n_in,
                              void* d_out, int out_size, void* d_ws, size_t ws_size,
                              hipStream_t stream) {
  (void)in_sizes; (void)n_in; (void)out_size; (void)ws_size;
  const float* x     = (const float*)d_in[0];
  const float* wq    = (const float*)d_in[2];
  const float* bq    = (const float*)d_in[3];
  const float* wk    = (const float*)d_in[4];
  const float* bk    = (const float*)d_in[5];
  const float* wv    = (const float*)d_in[6];
  const float* bv    = (const float*)d_in[7];
  const float* wvl   = (const float*)d_in[8];
  const float* bvl   = (const float*)d_in[9];
  const float* ab    = (const float*)d_in[10];
  const float* wth1  = (const float*)d_in[11];
  const float* bth1  = (const float*)d_in[12];
  const float* wth2  = (const float*)d_in[13];
  const float* bth2  = (const float*)d_in[14];
  const float* wproj = (const float*)d_in[15];
  const float* bproj = (const float*)d_in[16];
  const float* wfc1  = (const float*)d_in[17];
  const float* bfc1  = (const float*)d_in[18];
  const float* wmid  = (const float*)d_in[19];
  const float* bmid  = (const float*)d_in[20];
  const float* wfc2  = (const float*)d_in[21];
  const float* bfc2  = (const float*)d_in[22];
  const float* ls1   = (const float*)d_in[23];
  const float* ls2   = (const float*)d_in[24];
  const int*   bidx  = (const int*)d_in[25];
  float* out = (float*)d_out;

  char* p = (char*)d_ws;
  auto alloc = [&](size_t bytes) -> char* {
    char* r = p; p += (bytes + 255) & ~(size_t)255; return r;
  };
  float*    SC    = (float*)alloc(64 * sizeof(float));
  _Float16* WQKV  = (_Float16*)alloc((size_t)HIDD * CIN * 2);
  _Float16* WPROJ = (_Float16*)alloc((size_t)CIN * DHTOT * 2);
  _Float16* WFC1  = (_Float16*)alloc((size_t)HIDD * CIN * 2);
  _Float16* WFC2  = (_Float16*)alloc((size_t)CIN * HIDD * 2);
  float*    WTH1q = (float*)alloc(64 * 4);
  float*    WTH2q = (float*)alloc(64 * 4);
  float*    WVLq  = (float*)alloc((size_t)DHTOT * 9 * 4);
  float*    WMIDq = (float*)alloc((size_t)HIDD * 9 * 4);
  float*    BQKV  = (float*)alloc(HIDD * 4);
  _Float16* XH    = (_Float16*)alloc((size_t)BATCH * NP * CIN * 2);        // also T1 (b,49,384)
  _Float16* QKVR  = (_Float16*)alloc((size_t)BATCH * NP * HIDD * 2);       // QKNB+VBUF, later MID
  _Float16* RBIG  = (_Float16*)alloc((size_t)BATCH * NP * HIDD * 2);       // VL then H1
  _Float16* ATTN  = (_Float16*)alloc((size_t)BATCH * NHEADS * NP * NP * 2);// then X1
  _Float16* ATTN2 = (_Float16*)alloc((size_t)BATCH * NHEADS * NP * NP * 2);// then PROJ, F2
  _Float16* AV    = (_Float16*)alloc((size_t)BATCH * NP * DHTOT * 2);
  _Float16* T1   = XH;
  _Float16* QKNB = QKVR;                                   // (b,64,512) n-major q|k
  _Float16* VBUF = QKVR + (size_t)BATCH * NP * 512;        // (b,1024,64) channel-major v
  _Float16* MID  = QKVR;                                   // (b,64,1536) n-major
  _Float16* VL   = RBIG;                                   // (b,64,1024) n-major
  _Float16* H1   = RBIG;                                   // (b,64,1536) n-major
  _Float16* X1   = ATTN;                                   // (b,64,384)
  _Float16* PROJ = ATTN2;                                  // (b,64,384)
  _Float16* F2   = ATTN2;

  k_init<<<1, 64, 0, stream>>>(SC);

  k_quant_w_f16<<<NHKD,  256, 0, stream>>>(wq,    WQKV,                      CIN);
  k_quant_w_f16<<<NHKD,  256, 0, stream>>>(wk,    WQKV + (size_t)NHKD * CIN, CIN);
  k_quant_w_f16<<<DHTOT, 256, 0, stream>>>(wv,    WQKV + (size_t)512 * CIN,  CIN);
  k_quant_w_f16<<<CIN,   256, 0, stream>>>(wproj, WPROJ, DHTOT);
  k_quant_w_f16<<<HIDD,  256, 0, stream>>>(wfc1,  WFC1,  CIN);
  k_quant_w_f16<<<CIN,   256, 0, stream>>>(wfc2,  WFC2,  HIDD);
  k_quant_w_f32<<<8,     256, 0, stream>>>(wth1,  WTH1q, 8);
  k_quant_w_f32<<<8,     256, 0, stream>>>(wth2,  WTH2q, 8);
  k_quant_w_f32<<<DHTOT, 256, 0, stream>>>(wvl,   WVLq,  9);
  k_quant_w_f32<<<HIDD,  256, 0, stream>>>(wmid,  WMIDq, 9);
  k_concat_bias<<<6, 256, 0, stream>>>(bq, bk, bv, BQKV);

  k_pack_x<<<(BATCH * NP * CIN) / 256, 256, 0, stream>>>(x, XH);

  // qkv projection: q,k -> n-major QKNB (ostride 512), v -> channel-major VBUF
  k_gemm<<<dim3(HIDD / 64, BATCH), 128, 0, stream>>>(WQKV, XH, BQKV, QKNB, VBUF, SC,
                                                     HIDD, CIN, 512, -1, -2, 0);
  k_dw_vl<<<(BATCH * DHTOT * NSP + 255) / 256, 256, 0, stream>>>(VBUF, WVLq, bvl, VL, SC);
  k_qk<<<BATCH * NHEADS, 128, 0, stream>>>(QKNB, ab, bidx, ATTN, SC);
  k_headmix<<<(BATCH * NSP * NSP + 255) / 256, 256, 0, stream>>>(
      ATTN, ATTN2, WTH1q, bth1, SC, S_PRETH1, S_ATTN);
  k_softmax<<<(BATCH * NHEADS * NSP) / 8, 256, 0, stream>>>(ATTN2, ATTN, SC, S_ATTN);
  k_headmix<<<(BATCH * NSP * NSP + 255) / 256, 256, 0, stream>>>(
      ATTN, ATTN2, WTH2q, bth2, SC, -1, S_POST);
  k_av<<<dim3(BATCH * NHEADS, 2), 128, 0, stream>>>(ATTN2, VBUF, VL, AV, SC);
  k_relu_q<<<(BATCH * DHTOT * NSP + 255) / 256, 256, 0, stream>>>(AV, SC);
  k_gemm<<<dim3(CIN / 64, BATCH), 128, 0, stream>>>(WPROJ, AV, bproj, PROJ, nullptr, SC,
                                                    CIN, DHTOT, CIN, S_AVRELU, S_PROJ, 0);
  k_resid_a<<<(BATCH * NSP * CIN + 255) / 256, 256, 0, stream>>>(x, PROJ, ls1, T1, SC);
  k_resid_b<<<(BATCH * NP * CIN) / 256, 256, 0, stream>>>(T1, X1, SC);
  k_gemm<<<dim3(HIDD / 64, BATCH), 128, 0, stream>>>(WFC1, X1, bfc1, H1, nullptr, SC,
                                                     HIDD, CIN, HIDD, -1, S_H1, 1);
  k_dw_mid<<<(BATCH * HIDD * NSP + 255) / 256, 256, 0, stream>>>(H1, WMIDq, bmid, MID, SC);
  k_gemm<<<dim3(CIN / 64, BATCH), 128, 0, stream>>>(WFC2, MID, bfc2, F2, nullptr, SC,
                                                    CIN, HIDD, CIN, S_MID, S_F2, 0);
  k_final_a<<<(BATCH * NSP * CIN + 255) / 256, 256, 0, stream>>>(X1, F2, ls2, SC);
  k_final_b<<<(BATCH * NSP * CIN + 255) / 256, 256, 0, stream>>>(X1, F2, ls2, SC, out);
}
